// AgentLSTM_18202071401123
// MI455X (gfx1250) — matrix operand order repaired and checked by reading, not compiled, not run on hardware
//
#include <hip/hip_runtime.h>

// ---------------------------------------------------------------------------
// CDNA5 / gfx1250 implementation of the CNN+LSTM agent forward pass.
// All GEMM-like work (convs via implicit-GEMM, FC, LSTM gates) runs on
// v_wmma_f32_16x16x32_f16. Activations are f16 HWC; accumulation is f32.
// ---------------------------------------------------------------------------

typedef __attribute__((ext_vector_type(16))) _Float16 v16h;
typedef __attribute__((ext_vector_type(8)))  _Float16 v8h;
typedef __attribute__((ext_vector_type(8)))  float    v8f;

#define WMMA16(A, B, C) \
  __builtin_amdgcn_wmma_f32_16x16x32_f16(false, (A), false, (B), (short)0, (C), false, false)

__device__ __forceinline__ v8f zero8f() {
  v8f z;
#pragma unroll
  for (int i = 0; i < 8; i++) z[i] = 0.0f;
  return z;
}
__device__ __forceinline__ v8h zero8h() {
  v8h z;
#pragma unroll
  for (int i = 0; i < 8; i++) z[i] = (_Float16)0.0f;
  return z;
}
__device__ __forceinline__ v8h load8(const _Float16* p) { return *(const v8h*)p; }
__device__ __forceinline__ float sg(float x) { return 1.0f / (1.0f + __expf(-x)); }

// ---------------------------------------------------------------------------
// Preprocess: x (f32, NHWC, already HWC per image) -> f16, scaled by 1/255.
// ---------------------------------------------------------------------------
__global__ void k_prep(const float* __restrict__ x, _Float16* __restrict__ o, int n) {
  int i = blockIdx.x * 256 + threadIdx.x;
  if (i < n) o[i] = (_Float16)(x[i] * (1.0f / 255.0f));
}

// ---------------------------------------------------------------------------
// Pack conv weights (OIHW f32) into per-lane WMMA A-fragment layout.
// A 16x32 f16 mapping: lane L row M=L%16; element e ->
//   K = e + 16*(L>=16).  Global k = kb*32+K; kpos=k/CIN, cin=k%CIN.
// ---------------------------------------------------------------------------
__global__ void k_pack_conv(const float* __restrict__ w, _Float16* __restrict__ ap,
                            int COUT, int CIN, int total) {
  int idx = blockIdx.x * 256 + threadIdx.x;
  if (idx >= total) return;
  int e = idx & 15, lane = (idx >> 4) & 31, rest = idx >> 9;
  int MT = COUT >> 4;
  int mt = rest % MT, kb = rest / MT;
  int K = e + 16 * ((lane >> 4) & 1);
  int gk = kb * 32 + K;
  int kpos = gk / CIN, cin = gk % CIN;
  int cout = mt * 16 + (lane & 15);
  _Float16 v = (_Float16)0.0f;
  if (kpos < 9) v = (_Float16)w[(cout * CIN + cin) * 9 + kpos];
  ap[idx] = v;
}

// ---------------------------------------------------------------------------
// Pack fc_w [256,2048] f32 into B-fragment layout (B 32x16 f16: lane L col
// N=L%16, element e -> K = e + 16*(L>=16)).  Index: [kb(64)][nt(16)][lane][e].
// ---------------------------------------------------------------------------
__global__ void k_pack_fc(const float* __restrict__ fcw, _Float16* __restrict__ bp) {
  int idx = blockIdx.x * 256 + threadIdx.x;  // total 64*16*32*16 = 524288
  int e = idx & 15, lane = (idx >> 4) & 31;
  int nt = (idx >> 9) % 16, kb = (idx >> 9) / 16;
  int K = e + 16 * ((lane >> 4) & 1);
  int gk = kb * 32 + K;
  int n = nt * 16 + (lane & 15);
  bp[idx] = (_Float16)fcw[n * 2048 + gk];
}

// ---------------------------------------------------------------------------
// Pack [w_ih | w_hh] (each [1024,256] f32) into B-fragment layout:
// K = 0..511 (x-part then h-part), N = gate output j. [kb(16)][nt(64)][lane][e].
// ---------------------------------------------------------------------------
__global__ void k_pack_lstm(const float* __restrict__ w_ih, const float* __restrict__ w_hh,
                            _Float16* __restrict__ bp) {
  int idx = blockIdx.x * 256 + threadIdx.x;  // total 16*64*32*16 = 524288
  int e = idx & 15, lane = (idx >> 4) & 31;
  int nt = (idx >> 9) % 64, kb = (idx >> 9) / 64;
  int K = e + 8 * (e >= 8 ? 1 : 0) + 8 * ((lane >> 4) & 1);
  int gk = kb * 32 + K;
  int n = nt * 16 + (lane & 15);
  float v = (gk < 256) ? w_ih[n * 256 + gk] : w_hh[n * 256 + (gk - 256)];
  bp[idx] = (_Float16)v;
}

// ---------------------------------------------------------------------------
// Implicit-GEMM 3x3 conv, pad=1, stride=1, on WMMA f32_16x16x32_f16.
// in/out: [img][H][W][C] f16 (HWC).  One wave = one image x one 16-pixel tile,
// computing all COUT channels.  NK = ceil(9*CIN/32) K-blocks.
// ---------------------------------------------------------------------------
template <int CIN, int COUT, int W, bool PRELU, bool RES>
__global__ void k_conv(const _Float16* __restrict__ in, const _Float16* __restrict__ ap,
                       const float* __restrict__ bias, const _Float16* __restrict__ res,
                       _Float16* __restrict__ out) {
  constexpr int NK = (9 * CIN + 31) / 32;
  constexpr int MT = COUT / 16;
  constexpr int PT = (W * W) / 16;
  int gw = (int)((blockIdx.x * blockDim.x + threadIdx.x) >> 5);
  int lane = threadIdx.x & 31;
  int img = gw / PT, pt = gw - img * PT;
  if (img >= 2048) return;
  int p = pt * 16 + (lane & 15);
  int py = p / W, px = p - py * W;
  int hsel = (lane >> 4) & 1;
  const _Float16* ib = in + (size_t)img * W * W * CIN;

  v8f acc[MT];
#pragma unroll
  for (int m = 0; m < MT; m++) acc[m] = zero8f();

#pragma unroll
  for (int kb = 0; kb < NK; kb++) {
    v16h b;
    if constexpr (CIN == 32) {
      int ky = kb / 3 - 1, kx = kb % 3 - 1;
      int yy = py + ky, xx = px + kx;
      v8h lo = zero8h(), hi = zero8h();
      if (yy >= 0 && yy < W && xx >= 0 && xx < W) {
        const _Float16* q = ib + (yy * W + xx) * 32 + hsel * 16;
        lo = load8(q);
        hi = load8(q + 8);
      }
#pragma unroll
      for (int i = 0; i < 8; i++) { b[i] = lo[i]; b[i + 8] = hi[i]; }
    } else if constexpr (CIN == 16) {
      int kp = 2 * kb + hsel;  // K = e + 16*hsel -> kpos = 2kb+hsel, cin = e
      v8h lo = zero8h(), hi = zero8h();
      if (kp < 9) {
        int ky = kp / 3 - 1, kx = kp % 3 - 1;
        int yy = py + ky, xx = px + kx;
        if (yy >= 0 && yy < W && xx >= 0 && xx < W) {
          const _Float16* q = ib + (yy * W + xx) * 16;
          lo = load8(q);
          hi = load8(q + 8);
        }
      }
#pragma unroll
      for (int i = 0; i < 8; i++) { b[i] = lo[i]; b[i + 8] = hi[i]; }
    } else {  // CIN == 3: whole 27-deep im2col column in one K-block
#pragma unroll
      for (int e = 0; e < 16; e++) {
        int K = e + 16 * hsel;
        _Float16 v = (_Float16)0.0f;
        if (K < 27) {
          int kp = K / 3, cin = K - kp * 3;
          int yy = py + kp / 3 - 1, xx = px + kp % 3 - 1;
          if (yy >= 0 && yy < W && xx >= 0 && xx < W)
            v = ib[(yy * W + xx) * 3 + cin];
        }
        b[e] = v;
      }
    }
    if constexpr (PRELU) {
#pragma unroll
      for (int i = 0; i < 16; i++)
        b[i] = b[i] > (_Float16)0.0f ? b[i] : (_Float16)0.0f;
    }
#pragma unroll
    for (int m = 0; m < MT; m++) {
      const _Float16* qa = ap + (size_t)(((kb * MT + m) * 32 + lane)) * 16;
      v16h a = *(const v16h*)qa;
      acc[m] = WMMA16(a, b, acc[m]);
    }
  }

  // Epilogue: D lane holds pixel N=lane%16, VGPR r -> cout r + 8*(lane>=16).
  int cb0 = hsel * 8;
#pragma unroll
  for (int m = 0; m < MT; m++) {
    int cbase = m * 16 + cb0;
    size_t ofs = (size_t)img * W * W * COUT + (size_t)p * COUT + cbase;
    v8h rv = zero8h();
    if constexpr (RES) rv = load8(res + ofs);
    v8h o;
#pragma unroll
    for (int r = 0; r < 8; r++) {
      float v = acc[m][r] + bias[cbase + r];
      if constexpr (RES) v += (float)rv[r];
      o[r] = (_Float16)v;
    }
    *(v8h*)(out + ofs) = o;
  }
}

// ---------------------------------------------------------------------------
// MaxPool 3x3, stride 2, pad 1, HWC f16.
// ---------------------------------------------------------------------------
template <int C, int IW>
__global__ void k_pool(const _Float16* __restrict__ in, _Float16* __restrict__ out) {
  constexpr int OW = IW / 2;
  int idx = blockIdx.x * 256 + threadIdx.x;
  int total = 2048 * OW * OW * C;
  if (idx >= total) return;
  int c = idx % C, r1 = idx / C;
  int ox = r1 % OW, r2 = r1 / OW;
  int oy = r2 % OW, img = r2 / OW;
  const _Float16* ib = in + (size_t)img * IW * IW * C;
  float mv = -3.0e38f;
#pragma unroll
  for (int dy = 0; dy < 3; dy++) {
#pragma unroll
    for (int dx = 0; dx < 3; dx++) {
      int iy = 2 * oy - 1 + dy, ix = 2 * ox - 1 + dx;
      if (iy >= 0 && iy < IW && ix >= 0 && ix < IW)
        mv = fmaxf(mv, (float)ib[(iy * IW + ix) * C + c]);
    }
  }
  out[idx] = (_Float16)mv;
}

// ---------------------------------------------------------------------------
// FC: hidden = relu(relu(flat) @ fc_w^T + fc_b).  flat k = c*64 + p maps to
// act[img][p][c].  One wave = 16 images x 64 hidden units (4 N-tiles).
// ---------------------------------------------------------------------------
__global__ void k_fc(const _Float16* __restrict__ act, const _Float16* __restrict__ bp,
                     const float* __restrict__ fcb, _Float16* __restrict__ hidden) {
  int gw = (int)((blockIdx.x * blockDim.x + threadIdx.x) >> 5);
  int lane = threadIdx.x & 31;
  int ng = gw & 3, mtile = gw >> 2;
  if (mtile >= 128) return;
  int hsel = (lane >> 4) & 1;
  int imgA = mtile * 16 + (lane & 15);
  v8f acc[4];
#pragma unroll
  for (int j = 0; j < 4; j++) acc[j] = zero8f();

  for (int kb = 0; kb < 64; kb++) {
    v16h a;
#pragma unroll
    for (int e = 0; e < 16; e++) {
      int K = e + 16 * hsel;
      int gk = kb * 32 + K;
      int cch = gk >> 6, pp = gk & 63;  // k = c*64 + p
      _Float16 v = act[(size_t)imgA * 2048 + pp * 32 + cch];
      v = v > (_Float16)0.0f ? v : (_Float16)0.0f;  // relu(flat)
      a[e] = v;
    }
#pragma unroll
    for (int j = 0; j < 4; j++) {
      int nt = ng * 4 + j;
      v16h b = *(const v16h*)(bp + (size_t)((kb * 16 + nt) * 32 + lane) * 16);
      acc[j] = WMMA16(a, b, acc[j]);
    }
  }
#pragma unroll
  for (int j = 0; j < 4; j++) {
    int n = (ng * 4 + j) * 16 + (lane & 15);
    float bn = fcb[n];
#pragma unroll
    for (int r = 0; r < 8; r++) {
      int imgl = mtile * 16 + r + 8 * hsel;
      float v = acc[j][r] + bn;
      v = v > 0.0f ? v : 0.0f;
      hidden[(size_t)imgl * 256 + n] = (_Float16)v;
    }
  }
}

// ---------------------------------------------------------------------------
// LSTM: single persistent workgroup (1024 thr = 32 waves).  Per step one
// K=512 GEMM: gates[32x1024] = [x_t | h] @ [w_ih | w_hh]^T.  Each wave owns a
// fixed 16-batch x 16-gate-column slab and accumulates its i/f/g/o tiles
// together, so c stays in registers for all 64 steps; h round-trips via LDS.
// ---------------------------------------------------------------------------
__global__ void __launch_bounds__(1024) k_lstm(
    const _Float16* __restrict__ hidden, const _Float16* __restrict__ bp,
    const int* __restrict__ done, const float* __restrict__ h0,
    const float* __restrict__ c0, const float* __restrict__ b_ih,
    const float* __restrict__ b_hh, float* __restrict__ out) {
  __shared__ _Float16 hl[32 * 256];
  int tid = threadIdx.x, lane = tid & 31, wv = tid >> 5;
  int mt = wv & 1, jt = wv >> 1;  // jt in [0,16)
  for (int i = tid; i < 32 * 256; i += 1024) hl[i] = (_Float16)h0[i];
  int hsel = (lane >> 4) & 1;
  int j = jt * 16 + (lane & 15);
  int rbase = mt * 16 + hsel * 8;
  int arow = mt * 16 + (lane & 15);
  float cr[8];
#pragma unroll
  for (int r = 0; r < 8; r++) cr[r] = c0[(rbase + r) * 256 + j];
  float bi = b_ih[j] + b_hh[j];
  float bf = b_ih[256 + j] + b_hh[256 + j];
  float bg = b_ih[512 + j] + b_hh[512 + j];
  float bo = b_ih[768 + j] + b_hh[768 + j];
  __syncthreads();

  for (int t = 0; t < 64; t++) {
    float am = 1.0f - (float)done[t * 32 + arow];  // mask for this A row's h
    v8f ai = zero8f(), af = zero8f(), ag = zero8f(), ao = zero8f();
    for (int kb = 0; kb < 16; kb++) {
      int g0 = kb * 32 + 8 * hsel;  // A map: e<8 -> K=g0+e ; e>=8 -> K=g0+16+e-8
      v8h lo, hi;
      if (kb < 8) {  // x_t part from global hidden
        const _Float16* q = hidden + (size_t)(t * 32 + arow) * 256 + g0;
        lo = load8(q);
        hi = load8(q + 16);
      } else {  // h part from LDS, masked by (1-done)
        const _Float16* q = hl + arow * 256 + (g0 - 256);
        lo = load8(q);
        hi = load8(q + 16);
#pragma unroll
        for (int i = 0; i < 8; i++) {
          lo[i] = (_Float16)((float)lo[i] * am);
          hi[i] = (_Float16)((float)hi[i] * am);
        }
      }
      v16h a;
#pragma unroll
      for (int i = 0; i < 8; i++) { a[i] = lo[i]; a[i + 8] = hi[i]; }
      v16h b0 = *(const v16h*)(bp + (size_t)((kb * 64 + (0 * 16 + jt)) * 32 + lane) * 16);
      v16h b1 = *(const v16h*)(bp + (size_t)((kb * 64 + (1 * 16 + jt)) * 32 + lane) * 16);
      v16h b2 = *(const v16h*)(bp + (size_t)((kb * 64 + (2 * 16 + jt)) * 32 + lane) * 16);
      v16h b3 = *(const v16h*)(bp + (size_t)((kb * 64 + (3 * 16 + jt)) * 32 + lane) * 16);
      ai = WMMA16(a, b0, ai);
      af = WMMA16(a, b1, af);
      ag = WMMA16(a, b2, ag);
      ao = WMMA16(a, b3, ao);
    }
    __syncthreads();  // all h reads for this step done
#pragma unroll
    for (int r = 0; r < 8; r++) {
      int b = rbase + r;
      float m = 1.0f - (float)done[t * 32 + b];
      float iv = sg(ai[r] + bi);
      float fv = sg(af[r] + bf);
      float gv = tanhf(ag[r] + bg);
      float ov = sg(ao[r] + bo);
      float cn = fv * (cr[r] * m) + iv * gv;
      cr[r] = cn;
      float hv = ov * tanhf(cn);
      out[(size_t)(t * 32 + b) * 256 + j] = hv;
      hl[b * 256 + j] = (_Float16)hv;
    }
    __syncthreads();  // new h visible before next step's A gathers
  }
}

// ---------------------------------------------------------------------------
// Host-side orchestration.
// ---------------------------------------------------------------------------
extern "C" void kernel_launch(void* const* d_in, const int* in_sizes, int n_in,
                              void* d_out, int out_size, void* d_ws, size_t ws_size,
                              hipStream_t stream) {
  const float* x    = (const float*)d_in[0];
  const int*   done = (const int*)d_in[1];
  const float* h0   = (const float*)d_in[2];
  const float* c0   = (const float*)d_in[3];
  const float* cw0  = (const float*)d_in[4];
  const float* cb0  = (const float*)d_in[5];
  const float* rw0  = (const float*)d_in[6];
  const float* rb0  = (const float*)d_in[7];
  const float* cw1  = (const float*)d_in[8];
  const float* cb1  = (const float*)d_in[9];
  const float* rw1  = (const float*)d_in[10];
  const float* rb1  = (const float*)d_in[11];
  const float* cw2  = (const float*)d_in[12];
  const float* cb2  = (const float*)d_in[13];
  const float* rw2  = (const float*)d_in[14];
  const float* rb2  = (const float*)d_in[15];
  const float* fcw  = (const float*)d_in[16];
  const float* fcb  = (const float*)d_in[17];
  const float* w_ih = (const float*)d_in[18];
  const float* w_hh = (const float*)d_in[19];
  const float* b_ih = (const float*)d_in[20];
  const float* b_hh = (const float*)d_in[21];
  float* out = (float*)d_out;

  char* ws = (char*)d_ws;
  size_t off = 0;
  auto alloc = [&](size_t bytes) {
    size_t r = off;
    off = (off + bytes + 255) & ~(size_t)255;
    return r;
  };
  _Float16* x0h = (_Float16*)(ws + alloc(2048ull * 4096 * 3 * 2));    // 64x64x3
  _Float16* a1  = (_Float16*)(ws + alloc(2048ull * 4096 * 16 * 2));   // big ping (conv outs)
  _Float16* a2  = (_Float16*)(ws + alloc(2048ull * 1024 * 16 * 2));
  _Float16* a3  = (_Float16*)(ws + alloc(2048ull * 1024 * 16 * 2));
  _Float16* a4  = (_Float16*)(ws + alloc(2048ull * 1024 * 16 * 2));
  _Float16* hid = (_Float16*)(ws + alloc(2048ull * 256 * 2));
  // weight packs: NK*MT*512 halves each
  _Float16* p_c0 = (_Float16*)(ws + alloc(1ull * 1 * 512 * 2));
  _Float16* p_r0[4];
  for (int i = 0; i < 4; i++) p_r0[i] = (_Float16*)(ws + alloc(5ull * 1 * 512 * 2));
  _Float16* p_c1 = (_Float16*)(ws + alloc(5ull * 2 * 512 * 2));
  _Float16* p_c2 = (_Float16*)(ws + alloc(9ull * 2 * 512 * 2));
  _Float16* p_r1[4];
  for (int i = 0; i < 4; i++) p_r1[i] = (_Float16*)(ws + alloc(9ull * 2 * 512 * 2));
  _Float16* p_r2[4];
  for (int i = 0; i < 4; i++) p_r2[i] = (_Float16*)(ws + alloc(9ull * 2 * 512 * 2));
  _Float16* fcp = (_Float16*)(ws + alloc(64ull * 16 * 512 * 2));
  _Float16* lsp = (_Float16*)(ws + alloc(16ull * 64 * 512 * 2));

  // ---- weight packing ----
  auto packc = [&](const float* w, _Float16* ap, int COUT, int CIN) {
    int NK = (9 * CIN + 31) / 32;
    int total = NK * (COUT / 16) * 512;
    k_pack_conv<<<(total + 255) / 256, 256, 0, stream>>>(w, ap, COUT, CIN, total);
  };
  packc(cw0, p_c0, 16, 3);
  for (int i = 0; i < 4; i++) packc(rw0 + (size_t)i * 16 * 16 * 9, p_r0[i], 16, 16);
  packc(cw1, p_c1, 32, 16);
  packc(cw2, p_c2, 32, 32);
  for (int i = 0; i < 4; i++) packc(rw1 + (size_t)i * 32 * 32 * 9, p_r1[i], 32, 32);
  for (int i = 0; i < 4; i++) packc(rw2 + (size_t)i * 32 * 32 * 9, p_r2[i], 32, 32);
  k_pack_fc<<<2048, 256, 0, stream>>>(fcw, fcp);
  k_pack_lstm<<<2048, 256, 0, stream>>>(w_ih, w_hh, lsp);

  // ---- preprocess ----
  {
    int n = 2048 * 4096 * 3;
    k_prep<<<(n + 255) / 256, 256, 0, stream>>>(x, x0h, n);
  }

  // conv grid: one wave per (img, 16-pixel tile), 8 waves / block
  auto cg = [](int W) { return 2048 * (W * W / 16) / 8; };

  // ---- stage 0 ----
  k_conv<3, 16, 64, false, false><<<cg(64), 256, 0, stream>>>(x0h, p_c0, cb0, nullptr, a1);
  k_pool<16, 64><<<(2048 * 1024 * 16 + 255) / 256, 256, 0, stream>>>(a1, a2);
  k_conv<16, 16, 32, true, false><<<cg(32), 256, 0, stream>>>(a2, p_r0[0], rb0 + 0, nullptr, a3);
  k_conv<16, 16, 32, true, true><<<cg(32), 256, 0, stream>>>(a3, p_r0[1], rb0 + 16, a2, a4);
  k_conv<16, 16, 32, true, false><<<cg(32), 256, 0, stream>>>(a4, p_r0[2], rb0 + 32, nullptr, a3);
  k_conv<16, 16, 32, true, true><<<cg(32), 256, 0, stream>>>(a3, p_r0[3], rb0 + 48, a4, a2);

  // ---- stage 1 ----
  k_conv<16, 32, 32, false, false><<<cg(32), 256, 0, stream>>>(a2, p_c1, cb1, nullptr, a1);
  k_pool<32, 32><<<(2048 * 256 * 32 + 255) / 256, 256, 0, stream>>>(a1, a3);
  k_conv<32, 32, 16, true, false><<<cg(16), 256, 0, stream>>>(a3, p_r1[0], rb1 + 0, nullptr, a2);
  k_conv<32, 32, 16, true, true><<<cg(16), 256, 0, stream>>>(a2, p_r1[1], rb1 + 32, a3, a4);
  k_conv<32, 32, 16, true, false><<<cg(16), 256, 0, stream>>>(a4, p_r1[2], rb1 + 64, nullptr, a2);
  k_conv<32, 32, 16, true, true><<<cg(16), 256, 0, stream>>>(a2, p_r1[3], rb1 + 96, a4, a3);

  // ---- stage 2 ----
  k_conv<32, 32, 16, false, false><<<cg(16), 256, 0, stream>>>(a3, p_c2, cb2, nullptr, a1);
  k_pool<32, 16><<<(2048 * 64 * 32 + 255) / 256, 256, 0, stream>>>(a1, a2);
  k_conv<32, 32, 8, true, false><<<cg(8), 256, 0, stream>>>(a2, p_r2[0], rb2 + 0, nullptr, a3);
  k_conv<32, 32, 8, true, true><<<cg(8), 256, 0, stream>>>(a3, p_r2[1], rb2 + 32, a2, a4);
  k_conv<32, 32, 8, true, false><<<cg(8), 256, 0, stream>>>(a4, p_r2[2], rb2 + 64, nullptr, a3);
  k_conv<32, 32, 8, true, true><<<cg(8), 256, 0, stream>>>(a3, p_r2[3], rb2 + 96, a4, a2);

  // ---- head ----
  k_fc<<<64, 256, 0, stream>>>(a2, fcp, fcb, hid);
  k_lstm<<<1, 1024, 0, stream>>>(hid, lsp, done, h0, c0, b_ih, b_hh, out);
}
